// SE3FlowVectorField_77412490543136
// MI455X (gfx1250) — matrix operand order repaired and checked by reading, not compiled, not run on hardware
//
#include <hip/hip_runtime.h>
#include <stdint.h>

// ---------------------------------------------------------------------------
// Types for CDNA5 WMMA (wave32): v_wmma_f32_16x16x32_bf16
// ---------------------------------------------------------------------------
typedef __attribute__((ext_vector_type(16))) __bf16          v16bf;
typedef __attribute__((ext_vector_type(16))) unsigned short  v16u;
typedef __attribute__((ext_vector_type(8)))  float           v8f;
typedef __attribute__((ext_vector_type(4)))  unsigned int    su4;
typedef __attribute__((ext_vector_type(8)))  unsigned int    su8;

static __device__ __forceinline__ unsigned short f2bf(float f) {
  union { float f; unsigned u; } v; v.f = f;
  unsigned r = v.u + 0x7FFFu + ((v.u >> 16) & 1u);   // round-to-nearest-even
  return (unsigned short)(r >> 16);
}
static __device__ __forceinline__ float bf2f(unsigned short h) {
  union { unsigned u; float f; } v; v.u = ((unsigned)h) << 16;
  return v.f;
}
static __device__ __forceinline__ float silu_f(float x) {
  return x / (1.0f + __expf(-x));
}

// Load two 16-byte chunks (8 bf16 halves each) into one 32-byte fragment.
static __device__ __forceinline__ v16u ld2x16(const unsigned short* p0,
                                              const unsigned short* p1) {
  union { uint4 q[2]; v16u v; } u;
  u.q[0] = *(const uint4*)p0;
  u.q[1] = *(const uint4*)p1;
  return u.v;
}

// A-matrix 16x32 bf16 fragment (ISA 7.12.2): lane l holds row (l&15);
// halves 0..7 -> K = kt*32 + (l>>4)*8 .. +7 ; halves 8..15 -> +16.
static __device__ __forceinline__ v16u frag_a(const unsigned short* base,
                                              int rowBase, int stride,
                                              int kt, int lane) {
  const unsigned short* r = base + (size_t)(rowBase + (lane & 15)) * stride;
  int k0 = kt * 32 + (lane >> 4) * 8;
  return ld2x16(r + k0, r + k0 + 16);
}

// B-matrix 32x16 bf16 fragment, W stored transposed [N][K]:
// lane l holds col n=(l&15); halves 0..7 -> K = kt*32 + (l>>4)*8 .. +7 ; halves 8..15 -> +16.
static __device__ __forceinline__ v16u frag_b(const unsigned short* baseT,
                                              int colBase, int stride,
                                              int kt, int lane) {
  const unsigned short* r = baseT + (size_t)(colBase + (lane & 15)) * stride;
  int k0 = kt * 32 + (lane >> 4) * 8;
  return ld2x16(r + k0, r + k0 + 16);
}

static __device__ __forceinline__ v8f wmma_bf16(v16u a, v16u b, v8f c) {
  return __builtin_amdgcn_wmma_f32_16x16x32_bf16(
      false, __builtin_bit_cast(v16bf, a),
      false, __builtin_bit_cast(v16bf, b),
      (short)0, c, false, false);
}

// ---------------------------------------------------------------------------
// CDNA5 Tensor Data Mover: DMA a 256x256 bf16 weight tile (row stride 256)
// from global into LDS at lds_off, inserting 16 B of padding after every
// 512 B row (pad_interval=128 dwords, pad_amount=4 dwords) -> LDS stride 528B.
// D# per ISA 8.3/8.4; issue once (call from a single wave), completion via
// s_wait_tensorcnt.
// ---------------------------------------------------------------------------
static __device__ __forceinline__ void tdm_stage_weight(unsigned lds_off,
                                                        const unsigned short* gsrc) {
  unsigned long long ga = (unsigned long long)(size_t)gsrc;
  unsigned ga_lo = __builtin_amdgcn_readfirstlane((unsigned)ga);
  unsigned ga_hi = __builtin_amdgcn_readfirstlane((unsigned)(ga >> 32));
  unsigned lo    = __builtin_amdgcn_readfirstlane(lds_off);
  su4 g0;
  g0[0] = 1u;                                   // count=1 (valid user D#)
  g0[1] = lo;                                   // lds_addr
  g0[2] = ga_lo;                                // global_addr[31:0]
  g0[3] = (ga_hi & 0x01FFFFFFu) | (2u << 30);   // global_addr[56:32], type=2
  su8 g1;
  g1[0] = (1u << 16)                            // data_size = 2 bytes
        | (1u << 20)                            // pad_enable
        | (6u << 22)                            // pad_interval = 128 dwords
        | (3u << 25);                           // pad_amount = 4 dwords
  g1[1] = 256u << 16;                           // tensor_dim0[15:0] @ [63:48]
  g1[2] = 256u << 16;                           // tensor_dim1[15:0] @ [95:80]
  g1[3] = 256u << 16;                           // tile_dim0 @ [127:112]
  g1[4] = 256u;                                 // tile_dim1 @ [143:128]
  g1[5] = 256u;                                 // tensor_dim0_stride[31:0]
  g1[6] = 0u;
  g1[7] = 0u;
  asm volatile("tensor_load_to_lds %0, %1" :: "s"(g0), "s"(g1) : "memory");
}
static __device__ __forceinline__ void tdm_wait() {
  __builtin_amdgcn_s_wait_tensorcnt(0);
}

// Per-lane async copy global -> LDS (ASYNCcnt path)
static __device__ __forceinline__ void async_copy_b128(unsigned lds_off,
                                                       const void* gaddr) {
  asm volatile("global_load_async_to_lds_b128 %0, %1, off"
               :: "v"(lds_off), "v"((unsigned long long)(size_t)gaddr)
               : "memory");
}
static __device__ __forceinline__ void async_wait() {
  asm volatile("s_wait_asynccnt 0x0" ::: "memory");
}

// ---------------------------------------------------------------------------
// Weight transpose + f32->bf16 convert:  dst[n*K + k] = src[k*N + n]
// ---------------------------------------------------------------------------
__global__ void __launch_bounds__(256)
cvt_wt(unsigned short* dst, const float* src, int K, int N) {
  int i = blockIdx.x * 256 + threadIdx.x;
  if (i >= K * N) return;
  int n = i / K, k = i - n * K;
  dst[i] = f2bf(src[k * N + n]);
}

// ---------------------------------------------------------------------------
// Input projection: h = [feats(7), t(1), ctx(128)] @ Wp(136x256) + bp
// ---------------------------------------------------------------------------
__global__ void __launch_bounds__(256)
input_proj_k(const float* feats, const float* t, const float* ctx,
             const float* W, const float* b,
             float* h, unsigned short* hbf) {
  __shared__ float sin_[136];
  int node = blockIdx.x;            // b*256 + n
  int bidx = node >> 8;
  int tid = threadIdx.x;
  if (tid < 7)        sin_[tid] = feats[node * 7 + tid];
  else if (tid == 7)  sin_[7]   = t[bidx];
  else if (tid < 136) sin_[tid] = ctx[bidx * 128 + (tid - 8)];
  __syncthreads();
  float acc = b[tid];
  for (int i = 0; i < 136; ++i) acc += sin_[i] * W[i * 256 + tid];
  h[node * 256 + tid] = acc;
  hbf[node * 256 + tid] = f2bf(acc);
}

// ---------------------------------------------------------------------------
// Generic node-side WMMA GEMM: C[M=2048,256] = act(A[M,K] @ Wt[256,K]^T + b)
// ---------------------------------------------------------------------------
__global__ void __launch_bounds__(256)
gemm_node(const unsigned short* __restrict__ A, int lda,
          const unsigned short* __restrict__ Bt,
          const float* bias, const float* resid,
          float* outF, unsigned short* outB, int ldo,
          int K, int act) {
  int lane = threadIdx.x & 31;
  int wave = threadIdx.x >> 5;
  int rowBase = blockIdx.x * 16;
  int ktn = K >> 5;
  v8f acc0 = {}; v8f acc1 = {};
  for (int kt = 0; kt < ktn; ++kt) {
    if (kt + 1 < ktn)   // stream weights through L2 (global_prefetch_b8)
      __builtin_prefetch(Bt + (size_t)(wave * 32) * K + (kt + 1) * 32, 0, 1);
    v16u a  = frag_a(A, rowBase, lda, kt, lane);          // reused for 2 tiles
    v16u b0 = frag_b(Bt, wave * 32,      K, kt, lane);
    v16u b1 = frag_b(Bt, wave * 32 + 16, K, kt, lane);
    acc0 = wmma_bf16(a, b0, acc0);
    acc1 = wmma_bf16(a, b1, acc1);
  }
  for (int half = 0; half < 2; ++half) {
    v8f acc = half ? acc1 : acc0;
    int col = wave * 32 + half * 16 + (lane & 15);
    float bv = bias ? bias[col] : 0.0f;
    for (int r = 0; r < 8; ++r) {
      int row = rowBase + r + 8 * (lane >> 4);            // C/D VGPR layout
      float v = acc[r] + bv;
      if (act) v = silu_f(v);
      if (resid) v += resid[(size_t)row * 256 + col];
      if (outF) outF[(size_t)row * 256 + col] = v;
      if (outB) outB[(size_t)row * ldo + col] = f2bf(v);
    }
  }
}

// ---------------------------------------------------------------------------
// EGNN edge kernel: one workgroup per (b, i).  j processed in 2 chunks of 128.
// LDS (~275 KB dynamic, CDNA5 WGP has 320 KB):
//   sW   [256][264] bf16  -- staged weight (TDM / async-to-LDS)
//   sM1c [128][264] bf16  -- silu(edge1) chunk
//   sMc  [128][264] bf16  -- silu(edge2) chunk
// Per chunk: TDM-stage We2^T overlapped with M1 build -> WMMA GEMM1 ->
// async-stage Wc1^T overlapped with agg reduction -> WMMA GEMM2 -> wsc.
// ---------------------------------------------------------------------------
#define EST 264   // padded LDS row stride (bf16 elems); 528 B

__global__ void __launch_bounds__(256)
egnn_edge(const float* __restrict__ x, float* __restrict__ xnew,
          const float* __restrict__ Ai, const float* __restrict__ Aj,
          const unsigned short* __restrict__ hbf, unsigned short* __restrict__ hagg,
          const float* e1b, const float* wd,
          const unsigned short* __restrict__ We2t, const float* e2b,
          const unsigned short* __restrict__ Wc1t, const float* c1b,
          const float* c2w, const float* c2b) {
  extern __shared__ unsigned char smem[];
  unsigned short* sW   = (unsigned short*)smem;      // 256*EST
  unsigned short* sM1c = sW + 256 * EST;             // 128*EST
  unsigned short* sMc  = sM1c + 128 * EST;           // 128*EST
  float* sD2 = (float*)(sMc + 128 * EST);            // 256
  float* sAi = sD2 + 256;                            // 256
  float* sWd = sAi + 256;                            // 256
  float* sB1 = sWd + 256;                            // 256
  float* wsc = sB1 + 256;                            // 256
  float* xr  = wsc + 256;                            // 4

  int tid  = threadIdx.x;
  int lane = tid & 31, wave = tid >> 5;
  int wg = blockIdx.x;           // b*256 + i
  int i  = wg & 255;
  int nodeBase = wg & ~255;
  unsigned sW_off = (unsigned)(size_t)(void*)sW;     // LDS byte address

  float xi0 = x[wg * 3 + 0], xi1 = x[wg * 3 + 1], xi2 = x[wg * 3 + 2];
  {
    int j = tid;
    float dx = xi0 - x[(nodeBase + j) * 3 + 0];
    float dy = xi1 - x[(nodeBase + j) * 3 + 1];
    float dz = xi2 - x[(nodeBase + j) * 3 + 2];
    sD2[j]  = dx * dx + dy * dy + dz * dz;
    sAi[tid] = Ai[(size_t)wg * 256 + tid];
    sWd[tid] = wd[tid];
    sB1[tid] = e1b[tid];
    wsc[tid] = 0.0f;
    if (tid < 4) xr[tid] = 0.0f;
    hagg[(size_t)wg * 512 + tid] = hbf[(size_t)wg * 256 + tid];
  }
  __syncthreads();

  float aggReg = 0.0f;                 // agg_i[tid] accumulated over chunks

  for (int c = 0; c < 2; ++c) {
    int jb = c << 7;

    // TDM: DMA We2^T into sW (one instruction; overlaps M1 build)
    if (wave == 0) tdm_stage_weight(sW_off, We2t);

    // M1 chunk = silu(edge1) via factored form
    for (int idx = tid; idx < 128 * 256; idx += 256) {
      int jl = idx >> 8, k = idx & 255;
      float v = sAi[k] + Aj[(size_t)(nodeBase + jb + jl) * 256 + k]
              + sD2[jb + jl] * sWd[k] + sB1[k];
      sM1c[jl * EST + k] = f2bf(silu_f(v));
    }
    if (wave == 0) tdm_wait();
    __syncthreads();

    // GEMM1: sMc = silu(sM1c @ We2^T + b2); 128 tiles, 16 per wave
    for (int t = wave; t < 128; t += 8) {
      int tm = t >> 4, tn = t & 15;
      v8f acc = {};
      for (int kt = 0; kt < 8; ++kt) {
        v16u a = frag_a(sM1c, tm * 16, EST, kt, lane);   // ds_load
        v16u b = frag_b(sW,   tn * 16, EST, kt, lane);   // ds_load
        acc = wmma_bf16(a, b, acc);
      }
      int col = tn * 16 + (lane & 15);
      float bv = e2b[col];
      for (int r = 0; r < 8; ++r) {
        int row = tm * 16 + r + 8 * (lane >> 4);
        sMc[row * EST + col] = f2bf(silu_f(acc[r] + bv));
      }
    }
    __syncthreads();

    // async-stage Wc1^T into sW (ASYNCcnt path), overlapped with agg reduce
    for (int cc = 0; cc < 32; ++cc) {
      int chunk = tid + cc * 256;                  // 8192 x 16B chunks
      int row = chunk >> 5, col = chunk & 31;
      async_copy_b128(sW_off + (unsigned)(row * (EST * 2) + col * 16),
                      Wc1t + (size_t)row * 256 + col * 8);
    }
    for (int jl = 0; jl < 128; ++jl) {
      float m = bf2f(sMc[jl * EST + tid]);
      aggReg += ((jb + jl) == i) ? 0.0f : m;
    }
    async_wait();
    __syncthreads();

    // GEMM2: P = silu(sMc @ Wc1^T + bc1); fold coord2 dot into wsc
    for (int t = wave; t < 128; t += 8) {
      int tm = t >> 4, tn = t & 15;
      v8f acc = {};
      for (int kt = 0; kt < 8; ++kt) {
        v16u a = frag_a(sMc, tm * 16, EST, kt, lane);
        v16u b = frag_b(sW,  tn * 16, EST, kt, lane);
        acc = wmma_bf16(a, b, acc);
      }
      int col = tn * 16 + (lane & 15);
      float bv = c1b[col];
      float cw = c2w[col];
      for (int r = 0; r < 8; ++r) {
        int row = jb + tm * 16 + r + 8 * (lane >> 4);
        atomicAdd(&wsc[row], silu_f(acc[r] + bv) * cw);  // ds_add_f32
      }
    }
    __syncthreads();
  }

  // agg_i -> hagg second half (bf16)
  hagg[(size_t)wg * 512 + 256 + tid] = f2bf(aggReg);

  // coordinate update: x_i += sum_{j!=i} (x_i - x_j) * w_j / (N-1)
  {
    int j = tid;
    float w = wsc[j] + c2b[0];
    if (j == i) w = 0.0f;
    float dx = xi0 - x[(nodeBase + j) * 3 + 0];
    float dy = xi1 - x[(nodeBase + j) * 3 + 1];
    float dz = xi2 - x[(nodeBase + j) * 3 + 2];
    atomicAdd(&xr[0], dx * w);
    atomicAdd(&xr[1], dy * w);
    atomicAdd(&xr[2], dz * w);
  }
  __syncthreads();
  if (tid < 3)
    xnew[wg * 3 + tid] = x[wg * 3 + tid] + xr[tid] * (1.0f / 255.0f);
}

// ---------------------------------------------------------------------------
// Output heads: feature(7) | coord(3) | label(10), concatenated in d_out
// ---------------------------------------------------------------------------
__global__ void __launch_bounds__(256)
heads_k(const float* __restrict__ h,
        const float* fw, const float* fb,
        const float* cw, const float* cb,
        const float* lw, const float* lb,
        float* out) {
  __shared__ float sh[256];
  int node = blockIdx.x;
  int tid = threadIdx.x;
  sh[tid] = h[(size_t)node * 256 + tid];
  __syncthreads();
  if (tid < 20) {
    const float* W; const float* B; int O, o; size_t off;
    if (tid < 7)       { W = fw; B = fb; O = 7;  o = tid;      off = (size_t)node * 7 + o; }
    else if (tid < 10) { W = cw; B = cb; O = 3;  o = tid - 7;  off = 2048u * 7 + (size_t)node * 3 + o; }
    else               { W = lw; B = lb; O = 10; o = tid - 10; off = 2048u * 7 + 2048u * 3 + (size_t)node * 10 + o; }
    float acc = B[o];
    for (int k = 0; k < 256; ++k) acc += sh[k] * W[k * O + o];
    out[off] = acc;
  }
}

// ---------------------------------------------------------------------------
// Host launch. Param leaf order = jax tree flatten (dict keys sorted).
// ---------------------------------------------------------------------------
extern "C" void kernel_launch(void* const* d_in, const int* in_sizes, int n_in,
                              void* d_out, int out_size, void* d_ws, size_t ws_size,
                              hipStream_t stream) {
  (void)in_sizes; (void)n_in; (void)out_size; (void)ws_size;
  const int NODES = 2048;   // B*N = 8*256
  const int D = 256, L = 4;

  size_t off = 0;
  auto wsAlloc = [&](size_t bytes) -> void* {
    off = (off + 255) & ~((size_t)255);
    void* p = (char*)d_ws + off;
    off += bytes;
    return p;
  };

  float*          h    = (float*)wsAlloc((size_t)NODES * D * 4);
  unsigned short* hbf  = (unsigned short*)wsAlloc((size_t)NODES * D * 2);
  float*          Ai   = (float*)wsAlloc((size_t)NODES * D * 4);
  float*          Aj   = (float*)wsAlloc((size_t)NODES * D * 4);
  unsigned short* hagg = (unsigned short*)wsAlloc((size_t)NODES * 2 * D * 2);
  unsigned short* nh   = (unsigned short*)wsAlloc((size_t)NODES * D * 2);
  float*          x0   = (float*)wsAlloc((size_t)NODES * 3 * 4);
  float*          x1   = (float*)wsAlloc((size_t)NODES * 3 * 4);
  unsigned short *Whi[4], *Whj[4], *We2[4], *Wc1[4], *Wn1[4], *Wn2[4];
  for (int l = 0; l < L; ++l) {
    Whi[l] = (unsigned short*)wsAlloc(65536 * 2);
    Whj[l] = (unsigned short*)wsAlloc(65536 * 2);
    We2[l] = (unsigned short*)wsAlloc(65536 * 2);
    Wc1[l] = (unsigned short*)wsAlloc(65536 * 2);
    Wn1[l] = (unsigned short*)wsAlloc(131072 * 2);
    Wn2[l] = (unsigned short*)wsAlloc(65536 * 2);
  }

  const float* feats = (const float*)d_in[0];
  const float* coords = (const float*)d_in[1];
  const float* t_in = (const float*)d_in[2];
  const float* ctx = (const float*)d_in[3];
  const float* ch_b = (const float*)d_in[4];
  const float* ch_w = (const float*)d_in[5];
  const float* fh_b = (const float*)d_in[6];
  const float* fh_w = (const float*)d_in[7];
  const float* ip_b = (const float*)d_in[8];
  const float* ip_w = (const float*)d_in[9];
  const float* lh_b = (const float*)d_in[10];
  const float* lh_w = (const float*)d_in[11];

  // Pre-transpose weights to bf16 [N][K] for contiguous B fragments
  for (int l = 0; l < L; ++l) {
    int base = 12 + 12 * l;
    const float* e1w = (const float*)d_in[base + 5];
    const float* e2w = (const float*)d_in[base + 7];
    const float* c1w = (const float*)d_in[base + 1];
    const float* n1w = (const float*)d_in[base + 9];
    const float* n2w = (const float*)d_in[base + 11];
    cvt_wt<<<256, 256, 0, stream>>>(Whi[l], e1w,         256, 256);
    cvt_wt<<<256, 256, 0, stream>>>(Whj[l], e1w + 65536, 256, 256);
    cvt_wt<<<256, 256, 0, stream>>>(We2[l], e2w,         256, 256);
    cvt_wt<<<256, 256, 0, stream>>>(Wc1[l], c1w,         256, 256);
    cvt_wt<<<512, 256, 0, stream>>>(Wn1[l], n1w,         512, 256);
    cvt_wt<<<256, 256, 0, stream>>>(Wn2[l], n2w,         256, 256);
  }

  // Input projection
  input_proj_k<<<NODES, 256, 0, stream>>>(feats, t_in, ctx, ip_w, ip_b, h, hbf);

  const size_t EDGE_SMEM = (size_t)256 * EST * 2      // sW
                         + (size_t)2 * 128 * EST * 2  // sM1c + sMc
                         + (256 * 5 + 4) * 4;

  const float* xin = coords;
  float* xbufs[2] = { x0, x1 };
  for (int l = 0; l < L; ++l) {
    int base = 12 + 12 * l;
    const float* c1b = (const float*)d_in[base + 0];
    const float* c2b = (const float*)d_in[base + 2];
    const float* c2w = (const float*)d_in[base + 3];
    const float* e1b = (const float*)d_in[base + 4];
    const float* e1w = (const float*)d_in[base + 5];
    const float* e2b = (const float*)d_in[base + 6];
    const float* n1b = (const float*)d_in[base + 8];
    const float* n2b = (const float*)d_in[base + 10];
    const float* wd  = e1w + 512 * 256;   // dist2 row of edge1.w

    // Ai = h @ W_hi ; Aj = h @ W_hj   (per-node factoring of edge1)
    gemm_node<<<NODES / 16, 256, 0, stream>>>(hbf, 256, Whi[l], nullptr, nullptr,
                                              Ai, nullptr, 256, 256, 0);
    gemm_node<<<NODES / 16, 256, 0, stream>>>(hbf, 256, Whj[l], nullptr, nullptr,
                                              Aj, nullptr, 256, 256, 0);

    float* xout = xbufs[l & 1];
    egnn_edge<<<NODES, 256, EDGE_SMEM, stream>>>(xin, xout, Ai, Aj, hbf, hagg,
                                                 e1b, wd, We2[l], e2b,
                                                 Wc1[l], c1b, c2w, c2b);
    xin = xout;

    // nh = silu([h|agg] @ node1 + b) ; h += nh @ node2 + b (in place)
    gemm_node<<<NODES / 16, 256, 0, stream>>>(hagg, 512, Wn1[l], n1b, nullptr,
                                              nullptr, nh, 256, 512, 1);
    gemm_node<<<NODES / 16, 256, 0, stream>>>(nh, 256, Wn2[l], n2b, h,
                                              h, hbf, 256, 256, 0);
  }

  heads_k<<<NODES, 256, 0, stream>>>(h, fh_w, fh_b, ch_w, ch_b, lh_w, lh_b,
                                     (float*)d_out);
}
